// ADRiskGNN_68247030333845
// MI455X (gfx1250) — matrix operand order repaired and checked by reading, not compiled, not run on hardware
//
#include <hip/hip_runtime.h>
#include <math.h>

// ---------------- CDNA5 WMMA / TDM types ----------------
typedef __attribute__((ext_vector_type(16))) __bf16 v16bf;
typedef __attribute__((ext_vector_type(8)))  float  v8f;
typedef __attribute__((ext_vector_type(4)))  unsigned int v4u;
typedef __attribute__((ext_vector_type(8)))  int v8i;
typedef __attribute__((ext_vector_type(4)))  int v4i;

#if defined(__gfx1250__) && __has_builtin(__builtin_amdgcn_tensor_load_to_lds) && \
    __has_builtin(__builtin_amdgcn_s_wait_tensorcnt)
#define USE_TDM 1
#endif

#define HDIM 128
#define NHEAD 4
#define HEADD 32
#define NLAYER 3
#define NREL 7
#define NTYPE 5
#define EDGES 40000

enum { GM_STORE = 0, GM_STORE_ELU = 1, GM_SCATTER = 2, GM_LN = 3 };

__device__ __forceinline__ unsigned short f2bf(float f) {
  unsigned int b = __float_as_uint(f);
  unsigned int r = (b + 0x7FFFu + ((b >> 16) & 1u)) >> 16;  // round-to-nearest-even
  return (unsigned short)r;
}
__device__ __forceinline__ float eluf(float x) { return x > 0.f ? x : (__expf(x) - 1.f); }
__device__ __forceinline__ float sigm(float x) { return 1.f / (1.f + __expf(-x)); }
// order-preserving float->uint key for atomicMax on floats
__device__ __forceinline__ unsigned int fkey(float f) {
  unsigned int b = __float_as_uint(f);
  return (b & 0x80000000u) ? ~b : (b | 0x80000000u);
}
__device__ __forceinline__ float fkeyinv(unsigned int k) {
  unsigned int b = (k & 0x80000000u) ? (k & 0x7FFFFFFFu) : ~k;
  return __uint_as_float(b);
}

// Convert weight [nmat][K][128] f32 -> [nmat][128][K] bf16 (n-major, GEMM-B layout)
__global__ void conv_w(const float* __restrict__ W, unsigned short* __restrict__ out,
                       int K, int nmat) {
  int t = blockIdx.x * 256 + threadIdx.x;
  int per = K * HDIM;
  if (t >= nmat * per) return;
  int m = t / per, rem = t - m * per;
  int k = rem >> 7, n = rem & 127;
  out[(size_t)m * per + n * K + k] = f2bf(W[t]);
}

// ---------------- generic bf16-WMMA GEMM ----------------
// C[M,128] = epilogue( gather(A) @ B + bias )
// A rows: seg0 = A0[idx0[row]] (128 wide) ; optional seg1 = A1[idx1[row]] (K=256 concat)
// Bt: [128][K] bf16 n-major in global (pre-converted); staged to LDS via TDM when available.
__global__ __launch_bounds__(256) void wmma_gemm(
    const float* __restrict__ A0, const int* __restrict__ idx0,
    const float* __restrict__ A1, const int* __restrict__ idx1,
    const unsigned short* __restrict__ Bt, const float* __restrict__ bias,
    float* __restrict__ C,
    const int* __restrict__ sidx, const float* __restrict__ cw,
    const float* __restrict__ resid, const float* __restrict__ lng,
    const float* __restrict__ lnb,
    int M, int K, int mode) {
  extern __shared__ char smem[];
  unsigned short* sA = (unsigned short*)smem;                        // [64][K] bf16
  unsigned short* sB = (unsigned short*)(smem + (size_t)64 * K * 2); // [128][K] bf16, n-major
  float* sStage = (float*)(smem + (size_t)64 * K * 2 + (size_t)128 * K * 2); // [64][132] (LN)
  __shared__ int sRI[64];
  __shared__ int sRI2[64];

  const int tid = threadIdx.x;
  const int rbase = blockIdx.x * 64;

  if (tid < 64) {
    int row = rbase + tid;
    int v0 = -1, v1 = -1;
    if (row < M) {
      v0 = idx0 ? idx0[row] : row;
      v1 = idx1 ? idx1[row] : row;
    }
    sRI[tid] = v0;
    sRI2[tid] = v1;
    // warm L2 for the gathered rows (global_prefetch_b8)
    if (v0 >= 0) __builtin_prefetch(A0 + (size_t)v0 * HDIM, 0, 0);
    if (A1 && v1 >= 0) __builtin_prefetch(A1 + (size_t)v1 * HDIM, 0, 0);
  }
  __syncthreads();

#ifdef USE_TDM
  // Stage B via Tensor Data Mover: one wave issues the DMA descriptor, DMA overlaps
  // with the VALU/VMEM staging of A below; TENSORcnt waited before the barrier.
  if (tid < 32) {
    unsigned int ldsoff = (unsigned int)(size_t)sB;              // LDS byte offset
    unsigned long long ga = (unsigned long long)(size_t)Bt;      // 57-bit global addr
    v4u g0 = {1u,                                                // count=1, user mode
              ldsoff,                                            // lds_addr
              (unsigned int)ga,                                  // global_addr[31:0]
              (unsigned int)((ga >> 32) & 0x01FFFFFFu) | 0x80000000u}; // addr[56:32]|type=2
    v8i g1 = {(int)0x00010000,             // data_size=2B, no multicast/pad/iterate
              (int)((K & 0xFFFF) << 16),   // tensor_dim0 = K (elements)
              (int)(HDIM << 16),           // tensor_dim1 = 128
              (int)((K & 0xFFFF) << 16),   // tile_dim0 = K
              (int)HDIM,                   // tile_dim1 = 128 (tile_dim2 = 0)
              (int)K,                      // tensor_dim0_stride = K
              0, 0};
    v4i gz = {0, 0, 0, 0};
#if __clang_major__ >= 23
    v8i gz8 = {0, 0, 0, 0, 0, 0, 0, 0};
    __builtin_amdgcn_tensor_load_to_lds(g0, g1, gz, gz, gz8, 0);
#else
    __builtin_amdgcn_tensor_load_to_lds(g0, g1, gz, gz, 0);
#endif
  }
#else
  // Fallback: vectorized copy of pre-converted bf16 B tile.
  for (int t = tid; t < (K * HDIM) / 8; t += 256)
    ((uint4*)sB)[t] = ((const uint4*)Bt)[t];
#endif

  // Stage A (gathered, f32 -> bf16). Row stride of sources is always 128.
  const int K0 = A1 ? 128 : K;
  for (int t = tid; t < 64 * K; t += 256) {
    int rr = t / K, c = t - rr * K;
    float v = 0.f;
    if (c < K0) {
      int ri = sRI[rr];
      if (ri >= 0) v = A0[(size_t)ri * HDIM + c];
    } else {
      int ri = sRI2[rr];
      if (ri >= 0) v = A1[(size_t)ri * HDIM + (c - 128)];
    }
    sA[rr * K + c] = f2bf(v);
  }
#ifdef USE_TDM
  if (tid < 32) __builtin_amdgcn_s_wait_tensorcnt(0);
#endif
  __syncthreads();

  const int wave = tid >> 5, lane = tid & 31;
  const int wr = wave & 3;   // row tile 0..3 (rows 16*wr..)
  const int cg = wave >> 2;  // col group 0..1 (tiles cg*4 .. cg*4+3)
  const int half = lane >> 4, l16 = lane & 15;

  v8f acc0 = {}, acc1 = {}, acc2 = {}, acc3 = {};
  const unsigned short* sArow = sA + (wr * 16 + l16) * K;

  for (int kt = 0; kt < K; kt += 32) {
    union { v16bf v; unsigned int u[8]; } a, b0, b1, b2, b3;
#pragma unroll
    for (int p = 0; p < 8; ++p) {  // A 16x32 bf16 fragment pairs
      int kf = ((p & 3) * 2) + ((p >> 2) * 16) + half * 8 + kt;
      a.u[p] = *(const unsigned int*)(sArow + kf);
    }
#pragma unroll
    for (int p = 0; p < 8; ++p) {  // B 32x16 bf16 fragment: same k order as A
      int kb = ((p & 3) * 2) + ((p >> 2) * 16) + half * 8 + kt;
      b0.u[p] = *(const unsigned int*)(sB + ((cg * 4 + 0) * 16 + l16) * K + kb);
      b1.u[p] = *(const unsigned int*)(sB + ((cg * 4 + 1) * 16 + l16) * K + kb);
      b2.u[p] = *(const unsigned int*)(sB + ((cg * 4 + 2) * 16 + l16) * K + kb);
      b3.u[p] = *(const unsigned int*)(sB + ((cg * 4 + 3) * 16 + l16) * K + kb);
    }
    acc0 = __builtin_amdgcn_wmma_f32_16x16x32_bf16(false, a.v, false, b0.v, (short)0, acc0, false, false);
    acc1 = __builtin_amdgcn_wmma_f32_16x16x32_bf16(false, a.v, false, b1.v, (short)0, acc1, false, false);
    acc2 = __builtin_amdgcn_wmma_f32_16x16x32_bf16(false, a.v, false, b2.v, (short)0, acc2, false, false);
    acc3 = __builtin_amdgcn_wmma_f32_16x16x32_bf16(false, a.v, false, b3.v, (short)0, acc3, false, false);
  }

  // C/D layout: lane<16 -> N=lane, VGPR j -> M=j ; lane>=16 -> M=j+8
  if (mode == GM_STORE || mode == GM_STORE_ELU) {
#pragma unroll
    for (int t4 = 0; t4 < 4; ++t4) {
      v8f acc = (t4 == 0) ? acc0 : (t4 == 1) ? acc1 : (t4 == 2) ? acc2 : acc3;
      int col = (cg * 4 + t4) * 16 + l16;
      float bv = bias ? bias[col] : 0.f;
#pragma unroll
      for (int j = 0; j < 8; ++j) {
        int row = rbase + wr * 16 + half * 8 + j;
        if (row < M) {
          float v = acc[j] + bv;
          if (mode == GM_STORE_ELU) v = eluf(v);
          C[(size_t)row * HDIM + col] = v;
        }
      }
    }
  } else if (mode == GM_SCATTER) {
    float sc = cw[0];
#pragma unroll
    for (int t4 = 0; t4 < 4; ++t4) {
      v8f acc = (t4 == 0) ? acc0 : (t4 == 1) ? acc1 : (t4 == 2) ? acc2 : acc3;
      int col = (cg * 4 + t4) * 16 + l16;
      float bv = bias ? bias[col] : 0.f;
#pragma unroll
      for (int j = 0; j < 8; ++j) {
        int row = rbase + wr * 16 + half * 8 + j;
        if (row < M) {
          int d = sidx[row];
          atomicAdd(&C[(size_t)d * HDIM + col], (acc[j] + bv) * sc);
        }
      }
    }
  } else {  // GM_LN: residual + LayerNorm + ELU, row-wise via LDS stage
#pragma unroll
    for (int t4 = 0; t4 < 4; ++t4) {
      v8f acc = (t4 == 0) ? acc0 : (t4 == 1) ? acc1 : (t4 == 2) ? acc2 : acc3;
      int col = (cg * 4 + t4) * 16 + l16;
#pragma unroll
      for (int j = 0; j < 8; ++j) {
        int rl = wr * 16 + half * 8 + j;
        sStage[rl * 132 + col] = acc[j];
      }
    }
    __syncthreads();
    if (tid < 64) {
      int row = rbase + tid;
      if (row < M) {
        float* st = sStage + tid * 132;
        float mu = 0.f;
        for (int c = 0; c < HDIM; ++c) {
          float y = resid[(size_t)row * HDIM + c] + st[c] + bias[c];
          st[c] = y;
          mu += y;
        }
        mu *= (1.f / HDIM);
        float var = 0.f;
        for (int c = 0; c < HDIM; ++c) {
          float d = st[c] - mu;
          var += d * d;
        }
        var *= (1.f / HDIM);
        float inv = rsqrtf(var + 1e-5f);
        for (int c = 0; c < HDIM; ++c) {
          float o = (st[c] - mu) * inv * lng[c] + lnb[c];
          C[(size_t)row * HDIM + c] = eluf(o);
        }
      }
    }
  }
}

// ---------------- small support kernels ----------------
__global__ void fill_f32(float* p, float v, int n) {
  int t = blockIdx.x * 256 + threadIdx.x;
  if (t < n) p[t] = v;
}
__global__ void fill_u32(unsigned int* p, unsigned int v, int n) {
  int t = blockIdx.x * 256 + threadIdx.x;
  if (t < n) p[t] = v;
}
__global__ void elu_inplace(float* p, int n) {
  int t = blockIdx.x * 256 + threadIdx.x;
  if (t < n) p[t] = eluf(p[t]);
}

// h0 = elu(x @ Win + b_in)   (tiny K, VALU)
__global__ void input_proj(const float* __restrict__ x, const float* __restrict__ Win,
                           const float* __restrict__ bin, float* __restrict__ hout,
                           int N, int F) {
  int t = blockIdx.x * 256 + threadIdx.x;
  if (t >= N * HDIM) return;
  int n = t >> 7, j = t & 127;
  float acc = bin[j];
  const float* xr = x + (size_t)n * F;
  for (int f = 0; f < F; ++f) acc += xr[f] * Win[f * HDIM + j];
  hout[t] = eluf(acc);
}

// per-layer combine weights: cw[r] = sigmoid(rel_imp)*softmax_within_dsttype
__global__ void calc_cw(const float* __restrict__ ri, float* __restrict__ cw) {
  if (threadIdx.x != 0 || blockIdx.x != 0) return;
  float rw[NREL];
  for (int r = 0; r < NREL; ++r) rw[r] = sigm(ri[r]);
  const int ga[4] = {6, 0, 2, 3};
  const int gb[4] = {-1, 1, 4, 5};
  for (int g = 0; g < 4; ++g) {
    int x = ga[g], y = gb[g];
    if (y < 0) {
      cw[x] = rw[x];
    } else {
      float m = fmaxf(rw[x], rw[y]);
      float ea = __expf(rw[x] - m), eb = __expf(rw[y] - m);
      float z = ea + eb;
      cw[x] = rw[x] * ea / z;
      cw[y] = rw[y] * eb / z;
    }
  }
}

// s[e,h] = (Q.K)*scale*ew*(1-sig(theta)); segment max via ordered-uint atomicMax
__global__ void attn_score(const float* __restrict__ Q, const float* __restrict__ Kb,
                           const int* __restrict__ dst, const float* __restrict__ ewr,
                           const float* __restrict__ thr, unsigned int* __restrict__ m,
                           float* __restrict__ sb, int n) {
  int t = blockIdx.x * 256 + threadIdx.x;
  if (t >= n * NHEAD) return;
  int e = t >> 2, hh = t & 3;
  const float4* q = (const float4*)(Q + (size_t)e * HDIM + hh * HEADD);
  const float4* k = (const float4*)(Kb + (size_t)e * HDIM + hh * HEADD);
  float acc = 0.f;
#pragma unroll
  for (int i = 0; i < 8; ++i) {
    float4 a = q[i], b = k[i];
    acc += a.x * b.x + a.y * b.y + a.z * b.z + a.w * b.w;
  }
  float s = acc * 0.17677669529663687f * ewr[e] * (1.f - sigm(thr[0]));
  sb[t] = s;
  atomicMax(&m[(size_t)dst[e] * NHEAD + hh], fkey(s));
}

__global__ void attn_exp(float* __restrict__ sb, const int* __restrict__ dst,
                         const unsigned int* __restrict__ m, float* __restrict__ z, int n) {
  int t = blockIdx.x * 256 + threadIdx.x;
  if (t >= n * NHEAD) return;
  int e = t >> 2, hh = t & 3;
  int d = dst[e];
  float ex = __expf(sb[t] - fkeyinv(m[(size_t)d * NHEAD + hh]));
  sb[t] = ex;
  atomicAdd(&z[(size_t)d * NHEAD + hh], ex);
}

// pm = v * p  (p = ex / (z[dst]+1e-10))
__global__ void attn_pm(const float* __restrict__ V, const float* __restrict__ sb,
                        const float* __restrict__ z, const int* __restrict__ dst,
                        float* __restrict__ pm, int n) {
  int t = blockIdx.x * 256 + threadIdx.x;
  if (t >= n * HDIM) return;
  int e = t >> 7, c = t & 127, hh = c >> 5;
  int d = dst[e];
  float p = sb[(size_t)e * NHEAD + hh] / (z[(size_t)d * NHEAD + hh] + 1e-10f);
  pm[t] = V[t] * p;
}

// per-node risk head: sigmoid(elu(h@Wr1+br1)@Wr2+br2)
__global__ __launch_bounds__(64) void risk_head(const float* __restrict__ h,
                                                const float* __restrict__ Wr1,
                                                const float* __restrict__ br1,
                                                const float* __restrict__ Wr2,
                                                const float* __restrict__ br2,
                                                float* __restrict__ out) {
  int n = blockIdx.x, j = threadIdx.x;
  const float* hr = h + (size_t)n * HDIM;
  float acc = br1[j];
  for (int k = 0; k < HDIM; ++k) acc += hr[k] * Wr1[k * 64 + j];
  float t = eluf(acc) * Wr2[j];
  __shared__ float red[64];
  red[j] = t;
  __syncthreads();
  for (int s = 32; s > 0; s >>= 1) {
    if (j < s) red[j] += red[j + s];
    __syncthreads();
  }
  if (j == 0) out[n] = sigm(red[0] + br2[0]);
}

// edge head final dot: sigmoid(E1 . We2 + be2), one wave per edge
__global__ __launch_bounds__(256) void edge_out(const float* __restrict__ E1,
                                                const float* __restrict__ We2,
                                                const float* __restrict__ be2,
                                                float* __restrict__ out, int n) {
  int wave = threadIdx.x >> 5, lane = threadIdx.x & 31;
  int e = blockIdx.x * 8 + wave;
  if (e >= n) return;
  float acc = 0.f;
  for (int c = lane; c < HDIM; c += 32) acc += E1[(size_t)e * HDIM + c] * We2[c];
  for (int off = 16; off > 0; off >>= 1) acc += __shfl_down(acc, off, 32);
  if (lane == 0) out[e] = sigm(acc + be2[0]);
}

// ---------------- host orchestration ----------------
static void launch_gemm(hipStream_t s, const float* A0, const int* i0, const float* A1,
                        const int* i1, const unsigned short* Bt, const float* bias, float* C,
                        const int* sidx, const float* cw, const float* resid,
                        const float* g, const float* b, int M, int K, int mode) {
  dim3 grid((M + 63) / 64), block(256);
  size_t sh = (size_t)64 * K * 2 + (size_t)128 * K * 2 + (mode == GM_LN ? 64 * 132 * 4 : 0);
  wmma_gemm<<<grid, block, sh, s>>>(A0, i0, A1, i1, Bt, bias, C, sidx, cw, resid, g, b, M, K, mode);
}
static inline dim3 g1(int n) { return dim3((n + 255) / 256); }

extern "C" void kernel_launch(void* const* d_in, const int* in_sizes, int n_in,
                              void* d_out, int out_size, void* d_ws, size_t ws_size,
                              hipStream_t stream) {
  (void)in_sizes; (void)n_in; (void)out_size; (void)ws_size;
  const float* xs[NTYPE] = {(const float*)d_in[0], (const float*)d_in[1], (const float*)d_in[2],
                            (const float*)d_in[3], (const float*)d_in[4]};
  const int* ei = (const int*)d_in[5];       // [7,2,E]
  const float* ew = (const float*)d_in[6];   // [7,E]
  const float* theta = (const float*)d_in[7];
  const float* Win[NTYPE] = {(const float*)d_in[8], (const float*)d_in[9], (const float*)d_in[10],
                             (const float*)d_in[11], (const float*)d_in[12]};
  const float* b_in = (const float*)d_in[13];
  const float* Wq = (const float*)d_in[14];
  const float* Wk = (const float*)d_in[15];
  const float* Wv = (const float*)d_in[16];
  const float* Wmsg = (const float*)d_in[17];
  const float* bmsg = (const float*)d_in[18];
  const float* rel_imp = (const float*)d_in[19];
  const float* Wo = (const float*)d_in[20];
  const float* bo = (const float*)d_in[21];
  const float* lng = (const float*)d_in[22];
  const float* lnb = (const float*)d_in[23];
  const float* Wr1 = (const float*)d_in[24];
  const float* br1 = (const float*)d_in[25];
  const float* Wr2 = (const float*)d_in[26];
  const float* br2 = (const float*)d_in[27];
  const float* We1 = (const float*)d_in[28];
  const float* be1 = (const float*)d_in[29];
  const float* We2 = (const float*)d_in[30];
  const float* be2 = (const float*)d_in[31];

  const int NC[NTYPE] = {60000, 20000, 40000, 40000, 10000};
  const int NF[NTYPE] = {22, 6, 14, 5, 9};
  const int NOFF[NTYPE] = {0, 60000, 80000, 120000, 160000};
  const int RSI[NREL] = {0, 1, 0, 0, 2, 4, 0};
  const int RDI[NREL] = {1, 1, 2, 3, 2, 3, 0};
  const int TOTN = 170000;
  const int WSZ = HDIM * HDIM;  // 16384

  // carve scratch
  char* p = (char*)d_ws;
  auto carve = [&](size_t bytes) {
    void* r = (void*)p;
    p += (bytes + 255) & ~(size_t)255;
    return r;
  };
  float* h = (float*)carve((size_t)TOTN * HDIM * 4);
  float* comb = (float*)carve((size_t)TOTN * HDIM * 4);
  float* bufQ = (float*)carve((size_t)EDGES * HDIM * 4);
  float* bufK = (float*)carve((size_t)EDGES * HDIM * 4);
  float* bufV = (float*)carve((size_t)EDGES * HDIM * 4);
  float* bufPM = (float*)carve((size_t)EDGES * HDIM * 4);
  float* sbuf = (float*)carve((size_t)EDGES * NHEAD * 4);
  unsigned int* mbuf = (unsigned int*)carve((size_t)60000 * NHEAD * 4);
  float* zbuf = (float*)carve((size_t)60000 * NHEAD * 4);
  float* cwbuf = (float*)carve(NREL * 4);
  unsigned short* wq_bt = (unsigned short*)carve((size_t)NLAYER * NREL * WSZ * 2);
  unsigned short* wk_bt = (unsigned short*)carve((size_t)NLAYER * NREL * WSZ * 2);
  unsigned short* wv_bt = (unsigned short*)carve((size_t)NLAYER * NREL * WSZ * 2);
  unsigned short* wm_bt = (unsigned short*)carve((size_t)NLAYER * NREL * WSZ * 2);
  unsigned short* wo_bt = (unsigned short*)carve((size_t)NLAYER * NTYPE * WSZ * 2);
  unsigned short* we1_bt = (unsigned short*)carve((size_t)256 * HDIM * 2);

  // one-time (per call) weight conversion to bf16 n-major
  conv_w<<<g1(NLAYER * NREL * WSZ), 256, 0, stream>>>(Wq, wq_bt, HDIM, NLAYER * NREL);
  conv_w<<<g1(NLAYER * NREL * WSZ), 256, 0, stream>>>(Wk, wk_bt, HDIM, NLAYER * NREL);
  conv_w<<<g1(NLAYER * NREL * WSZ), 256, 0, stream>>>(Wv, wv_bt, HDIM, NLAYER * NREL);
  conv_w<<<g1(NLAYER * NREL * WSZ), 256, 0, stream>>>(Wmsg, wm_bt, HDIM, NLAYER * NREL);
  conv_w<<<g1(NLAYER * NTYPE * WSZ), 256, 0, stream>>>(Wo, wo_bt, HDIM, NLAYER * NTYPE);
  conv_w<<<g1(256 * HDIM), 256, 0, stream>>>(We1, we1_bt, 256, 1);

  // input projections
  for (int i = 0; i < NTYPE; ++i)
    input_proj<<<g1(NC[i] * HDIM), 256, 0, stream>>>(xs[i], Win[i], b_in + i * HDIM,
                                                     h + (size_t)NOFF[i] * HDIM, NC[i], NF[i]);

  for (int l = 0; l < NLAYER; ++l) {
    calc_cw<<<1, 32, 0, stream>>>(rel_imp + l * NREL, cwbuf);
    fill_f32<<<g1(160000 * HDIM), 256, 0, stream>>>(comb, 0.f, 160000 * HDIM);

    for (int r = 0; r < NREL; ++r) {
      const int si = RSI[r], di = RDI[r];
      const int* srcI = ei + ((size_t)r * 2 + 0) * EDGES;
      const int* dstI = ei + ((size_t)r * 2 + 1) * EDGES;
      const size_t wrel = ((size_t)l * NREL + r);
      // Q = gather(h_di, dst) @ Wq ; K,V = gather(h_si, src) @ Wk/Wv   (WMMA bf16)
      launch_gemm(stream, h + (size_t)NOFF[di] * HDIM, dstI, nullptr, nullptr,
                  wq_bt + wrel * WSZ, nullptr, bufQ, nullptr, nullptr, nullptr, nullptr,
                  nullptr, EDGES, HDIM, GM_STORE);
      launch_gemm(stream, h + (size_t)NOFF[si] * HDIM, srcI, nullptr, nullptr,
                  wk_bt + wrel * WSZ, nullptr, bufK, nullptr, nullptr, nullptr, nullptr,
                  nullptr, EDGES, HDIM, GM_STORE);
      launch_gemm(stream, h + (size_t)NOFF[si] * HDIM, srcI, nullptr, nullptr,
                  wv_bt + wrel * WSZ, nullptr, bufV, nullptr, nullptr, nullptr, nullptr,
                  nullptr, EDGES, HDIM, GM_STORE);
      // segment softmax over dst
      const int nd = NC[di];
      fill_u32<<<g1(nd * NHEAD), 256, 0, stream>>>(mbuf, 0u, nd * NHEAD);
      fill_f32<<<g1(nd * NHEAD), 256, 0, stream>>>(zbuf, 0.f, nd * NHEAD);
      attn_score<<<g1(EDGES * NHEAD), 256, 0, stream>>>(bufQ, bufK, dstI, ew + (size_t)r * EDGES,
                                                        theta + r, mbuf, sbuf, EDGES);
      attn_exp<<<g1(EDGES * NHEAD), 256, 0, stream>>>(sbuf, dstI, mbuf, zbuf, EDGES);
      attn_pm<<<g1(EDGES * HDIM), 256, 0, stream>>>(bufV, sbuf, zbuf, dstI, bufPM, EDGES);
      // comb[dst] += ((pm @ Wmsg) + bmsg) * (rw * softmax_w)   (WMMA + atomic scatter)
      launch_gemm(stream, bufPM, nullptr, nullptr, nullptr, wm_bt + wrel * WSZ,
                  bmsg + wrel * HDIM, comb + (size_t)NOFF[di] * HDIM, dstI, cwbuf + r, nullptr,
                  nullptr, nullptr, EDGES, HDIM, GM_SCATTER);
    }
    // node update: h = elu(LN(h + comb@Wo + bo)*g + b)  (WMMA + fused LN epilogue)
    for (int i = 0; i < 4; ++i) {
      const size_t wli = (size_t)l * NTYPE + i;
      launch_gemm(stream, comb + (size_t)NOFF[i] * HDIM, nullptr, nullptr, nullptr,
                  wo_bt + wli * WSZ, bo + wli * HDIM, h + (size_t)NOFF[i] * HDIM, nullptr,
                  nullptr, h + (size_t)NOFF[i] * HDIM, lng + wli * HDIM, lnb + wli * HDIM, NC[i],
                  HDIM, GM_LN);
    }
    // AIAgent has no incoming relations: reference still re-applies ELU
    elu_inplace<<<g1(NC[4] * HDIM), 256, 0, stream>>>(h + (size_t)NOFF[4] * HDIM, NC[4] * HDIM);
  }

  // outputs: 5 node-risk vectors then 7 edge-probability vectors
  float* out = (float*)d_out;
  for (int i = 0; i < NTYPE; ++i)
    risk_head<<<dim3(NC[i]), 64, 0, stream>>>(h + (size_t)NOFF[i] * HDIM, Wr1, br1, Wr2, br2,
                                              out + NOFF[i]);
  for (int r = 0; r < NREL; ++r) {
    const int si = RSI[r], di = RDI[r];
    const int* srcI = ei + ((size_t)r * 2 + 0) * EDGES;
    const int* dstI = ei + ((size_t)r * 2 + 1) * EDGES;
    // E1 = elu([h_src | h_dst] @ We1 + be1)  — K=256 concat-gather WMMA GEMM
    launch_gemm(stream, h + (size_t)NOFF[si] * HDIM, srcI, h + (size_t)NOFF[di] * HDIM, dstI,
                we1_bt, be1, bufQ, nullptr, nullptr, nullptr, nullptr, nullptr, EDGES, 256,
                GM_STORE_ELU);
    edge_out<<<dim3((EDGES + 7) / 8), 256, 0, stream>>>(bufQ, We2, be2,
                                                        out + TOTN + (size_t)r * EDGES, EDGES);
  }
}
